// HGCN_77893526880286
// MI455X (gfx1250) — compile-verified, matrix operand order checked by reading, not run on hardware
//
#include <hip/hip_runtime.h>

// ---------------------------------------------------------------------------
// HGCN (2-layer hyperbolic GCN), N=16384, D=64, c=1.
//
// Dominant cost: adj_n @ logmap0(h) per layer = streaming 1 GiB f32 adj.
// Memory-bound target on MI455X: ~46us/layer at 23.3 TB/s. adj is converted
// f32->bf16 in-register (v_cvt_pk_bf16_f32 when available) and fed to
// v_wmma_f32_16x16x32_bf16 with f32 accumulation; bf16 WMMA throughput keeps
// this memory-bound, unlike the K=4 f32 WMMA path. The B operand (2.6 MB)
// stays L2-resident; adj loads are non-temporal so the stream does not evict
// it. Row sums are fused into the GEMM via an appended ones-column (output
// col 64), avoiding an extra 1 GiB pass over adj.
// ---------------------------------------------------------------------------

#define HN 16384          // rows/cols of adj
#define HD 64             // feature dim
#define NTB 80            // B columns: 64 features + ones col (64) + pad
#define LDT 16384         // TbT row stride (elements): TbT[n][k], k contiguous
#define SPLITK 4          // K-split for the big GEMM
#define KCHUNK (HN / SPLITK)

#define MIN_NORM 1e-15f
#define HEPS 1e-7f
#define MAX_NORM 1000000.0f

typedef __attribute__((ext_vector_type(4)))  float  v4f;   // true clang vector
typedef __attribute__((ext_vector_type(8)))  float  v8f;
typedef __attribute__((ext_vector_type(16))) __bf16 v16bf;
typedef __attribute__((ext_vector_type(8)))  __bf16 v8bf;
typedef __attribute__((ext_vector_type(2)))  __bf16 v2bf;

#if defined(__has_builtin)
#if __has_builtin(__builtin_amdgcn_cvt_pk_bf16_f32)
#define HGCN_HAVE_PK_CVT 1
#endif
#endif

static __device__ inline __bf16 f2bf(float f) {
  unsigned u;
  __builtin_memcpy(&u, &f, 4);
  unsigned r = (u + 0x7FFFu + ((u >> 16) & 1u)) >> 16;   // RNE
  unsigned short s = (unsigned short)r;
  __bf16 h;
  __builtin_memcpy(&h, &s, 2);
  return h;
}

// Convert a v4f into bf16x4 at positions [e..e+3] of a v16bf.
static __device__ inline void put4(v16bf& A, int e, v4f f) {
#if defined(HGCN_HAVE_PK_CVT)
  v2bf p0 = __builtin_amdgcn_cvt_pk_bf16_f32(f[0], f[1]);
  v2bf p1 = __builtin_amdgcn_cvt_pk_bf16_f32(f[2], f[3]);
  A[e + 0] = p0[0]; A[e + 1] = p0[1];
  A[e + 2] = p1[0]; A[e + 3] = p1[1];
#else
  // native fptrunc; falls back to VALU sequence if no cvt instruction
  A[e + 0] = (__bf16)f[0]; A[e + 1] = (__bf16)f[1];
  A[e + 2] = (__bf16)f[2]; A[e + 3] = (__bf16)f[3];
#endif
}

// ----- row-local hyperbolic helpers (c == 1, sqrtK == 1) --------------------

static __device__ inline float norm_tail(const float* v) {
  float s = 0.f;
  #pragma unroll
  for (int k = 1; k < HD; ++k) s = fmaf(v[k], v[k], s);
  return sqrtf(s);
}

// out = proj(expmap0(u)); uses only u[1:]
static __device__ inline void expmap0_proj(const float* u, float* out) {
  float xn = fmaxf(norm_tail(u), MIN_NORM);
  float scale = sinhf(xn) / xn;
  float s = 0.f;
  #pragma unroll
  for (int k = 1; k < HD; ++k) {
    float y = scale * u[k];
    out[k] = y;
    s = fmaf(y, y, s);
  }
  out[0] = sqrtf(fmaxf(1.0f + s, HEPS));   // proj: recompute time coord
}

// u = logmap0(x)
static __device__ inline void logmap0_f(const float* x, float* u) {
  float yn = fmaxf(norm_tail(x), MIN_NORM);
  float th = fmaxf(x[0], 1.0f + HEPS);
  float s = acoshf(th) / yn;
  u[0] = 0.f;
  #pragma unroll
  for (int k = 1; k < HD; ++k) u[k] = s * x[k];
}

// ---------------------------------------------------------------------------
// Prologue per layer (thread-per-row):
//   t = logmap0(x_hyp or h1); mv = t @ W^T; res = proj(expmap0(mv));
//   h = proj(mobius_add(res, hyp_bias)); tb = logmap0(h);
//   write TbT[n][i] = bf16(tb[n]), TbT[64][i] = 1, TbT[65..79][i] = 0.
// ---------------------------------------------------------------------------
__global__ __launch_bounds__(256) void hgcn_pre_layer(
    const float* __restrict__ in, const float* __restrict__ W,
    const float* __restrict__ b, __bf16* __restrict__ TbT, int first) {
  int i = blockIdx.x * blockDim.x + threadIdx.x;
  if (i >= HN) return;
  const float* row = in + (size_t)i * HD;

  float t[HD];
  {
    float tmp[HD];
    if (first) {
      float xh[HD];
      tmp[0] = 0.f;
      #pragma unroll
      for (int k = 1; k < HD; ++k) tmp[k] = row[k];   // proj_tan0(x)
      expmap0_proj(tmp, xh);                          // x_hyp
      logmap0_f(xh, t);
    } else {
      #pragma unroll
      for (int k = 0; k < HD; ++k) tmp[k] = row[k];   // h1 already on manifold
      logmap0_f(tmp, t);
    }
  }

  // mv = t @ W^T  (mv[o] = sum_k t[k] * W[o][k]); W is 16 KB, L2/K$-resident
  float mv[HD];
  for (int o = 0; o < HD; ++o) {
    const float* wr = W + (size_t)o * HD;
    float a = 0.f;
    #pragma unroll
    for (int k = 0; k < HD; ++k) a = fmaf(t[k], wr[k], a);
    mv[o] = a;
  }
  float res[HD];
  expmap0_proj(mv, res);

  // hyp_bias = proj(expmap0(proj_tan0(b)))
  float hb[HD];
  {
    float bb[HD];
    bb[0] = 0.f;
    #pragma unroll
    for (int k = 1; k < HD; ++k) bb[k] = b[k];
    expmap0_proj(bb, hb);
  }
  // u2 = logmap0(hyp_bias)
  float u2[HD];
  logmap0_f(hb, u2);

  // v = proj_tan(u2 - alpha*vvec, res)   [ptransp0]
  float yn = fmaxf(norm_tail(res), MIN_NORM);
  float alpha = 0.f;
  #pragma unroll
  for (int k = 1; k < HD; ++k) alpha = fmaf(res[k] / yn, u2[k], alpha);
  float w[HD];
  float fac = 1.0f - res[0];                  // (sqrtK - x0), sqrtK = 1
  w[0] = alpha * yn;                          // u2[0] - alpha*(-yn), u2[0]=0
  #pragma unroll
  for (int k = 1; k < HD; ++k) w[k] = u2[k] - alpha * (fac * res[k] / yn);
  float ux = 0.f;
  #pragma unroll
  for (int k = 1; k < HD; ++k) ux = fmaf(res[k], w[k], ux);
  w[0] = ux / fmaxf(res[0], HEPS);            // proj_tan

  // expmap(w, res) then proj
  float md = -w[0] * w[0];
  #pragma unroll
  for (int k = 1; k < HD; ++k) md = fmaf(w[k], w[k], md);
  float normu = fminf(sqrtf(fmaxf(md, HEPS)), MAX_NORM);
  float theta = fmaxf(normu, MIN_NORM);
  float ch = coshf(theta), shd = sinhf(theta) / theta;
  float h[HD];
  float s2 = 0.f;
  #pragma unroll
  for (int k = 1; k < HD; ++k) {
    float hv = fmaf(ch, res[k], shd * w[k]);
    h[k] = hv;
    s2 = fmaf(hv, hv, s2);
  }
  h[0] = sqrtf(fmaxf(1.0f + s2, HEPS));       // proj (idempotent wrt outer proj)

  // tb = logmap0(h); write transposed bf16 + ones column for fused rowsum
  float tb[HD];
  logmap0_f(h, tb);
  #pragma unroll
  for (int n = 0; n < HD; ++n) TbT[(size_t)n * LDT + i] = f2bf(tb[n]);
  TbT[(size_t)64 * LDT + i] = f2bf(1.0f);
  #pragma unroll
  for (int n = 65; n < NTB; ++n) TbT[(size_t)n * LDT + i] = f2bf(0.0f);
}

// ---------------------------------------------------------------------------
// Aggregation GEMM: partial[c][i][n] = sum_{k in chunk c} adj[i][k]*TbT[n][k]
// One wave per (16-row M tile, K chunk); 5 N-tiles (80 cols) in registers.
// adj f32 is streamed once per layer (non-temporal), converted to bf16
// in-register, fed to v_wmma_f32_16x16x32_bf16. B operand is L2-resident.
// ---------------------------------------------------------------------------
__global__ __launch_bounds__(256) void hgcn_agg_wmma(
    const float* __restrict__ adj, const __bf16* __restrict__ TbT,
    float* __restrict__ partial) {
  const int lane = threadIdx.x & 31;
  const int wid = (blockIdx.x * blockDim.x + threadIdx.x) >> 5;
  const int kchunk = wid & (SPLITK - 1);
  const int mtile = wid >> 2;
  if (mtile >= HN / 16) return;

  const int nn = lane & 15;        // column-in-tile (B/C) / row-in-tile (A)
  const int hi = lane >> 4;        // lane group selects K sub-range
  const int koff = hi * 8;

  const float* arow = adj + (size_t)(mtile * 16 + nn) * HN;

  v8f acc[5];
  #pragma unroll
  for (int nt = 0; nt < 5; ++nt) acc[nt] = (v8f){0.f,0.f,0.f,0.f,0.f,0.f,0.f,0.f};

  const int kbeg = kchunk * KCHUNK;
  const int kend = kbeg + KCHUNK;
  #pragma unroll 2
  for (int k0 = kbeg; k0 < kend; k0 += 32) {
    const int kb = k0 + koff;

    // A tile (16x32 f32 -> bf16): lane row = nn, K = kb..kb+7, kb+16..kb+23
    // Non-temporal: adj is a pure stream; keep it out of the reused L2 set.
    const v4f a0 = __builtin_nontemporal_load(
        reinterpret_cast<const v4f*>(arow + kb));
    const v4f a1 = __builtin_nontemporal_load(
        reinterpret_cast<const v4f*>(arow + kb + 4));
    const v4f a2 = __builtin_nontemporal_load(
        reinterpret_cast<const v4f*>(arow + kb + 16));
    const v4f a3 = __builtin_nontemporal_load(
        reinterpret_cast<const v4f*>(arow + kb + 20));
    __builtin_prefetch(arow + kb + 64, 0, 1);   // global_prefetch, 2 steps ahead

    v16bf A;
    put4(A, 0,  a0);
    put4(A, 4,  a1);
    put4(A, 8,  a2);
    put4(A, 12, a3);

    #pragma unroll
    for (int nt = 0; nt < 5; ++nt) {
      // B tile: lane col = nn, K contiguous in TbT rows -> 2x16B loads
      const v8bf* bp =
          reinterpret_cast<const v8bf*>(TbT + (size_t)(nt * 16 + nn) * LDT + kb);
      const v8bf blo = bp[0];        // K = kb..kb+7
      const v8bf bhi = bp[2];        // K = kb+16..kb+23
      v16bf B = __builtin_shufflevector(blo, bhi,
                                        0, 1, 2, 3, 4, 5, 6, 7,
                                        8, 9, 10, 11, 12, 13, 14, 15);
      acc[nt] = __builtin_amdgcn_wmma_f32_16x16x32_bf16(
          false, A, false, B, (short)0, acc[nt], false, false);
    }
  }

  // C layout: VGPR r holds (m = r + 8*hi, n = nn)
  float* pp = partial + ((size_t)kchunk * HN + (size_t)mtile * 16) * NTB;
  const int mhi = hi * 8;
  #pragma unroll
  for (int r = 0; r < 8; ++r) {
    #pragma unroll
    for (int nt = 0; nt < 5; ++nt) {
      pp[(size_t)(r + mhi) * NTB + nt * 16 + nn] = acc[nt][r];
    }
  }
}

// ---------------------------------------------------------------------------
// Epilogue per layer (thread-per-row): reduce split-K partials, apply fused
// 1/rowsum (col 64), then proj(expmap0(.)) and hyp_act -> layer output.
// ---------------------------------------------------------------------------
__global__ __launch_bounds__(256) void hgcn_post_agg(
    const float* __restrict__ partial, float* __restrict__ out) {
  int i = blockIdx.x * blockDim.x + threadIdx.x;
  if (i >= HN) return;

  float s = 0.f;
  #pragma unroll
  for (int c = 0; c < SPLITK; ++c)
    s += partial[((size_t)c * HN + i) * NTB + 64];
  float rinv = (s != 0.0f) ? 1.0f / s : 0.0f;

  float a[HD];
  #pragma unroll
  for (int n = 0; n < HD; ++n) {
    float acc = 0.f;
    #pragma unroll
    for (int c = 0; c < SPLITK; ++c)
      acc += partial[((size_t)c * HN + i) * NTB + n];
    a[n] = acc * rinv;
  }

  float h[HD];
  expmap0_proj(a, h);          // hyp_agg: proj(expmap0(adj_n @ logmap0))
  float z[HD];
  logmap0_f(h, z);             // hyp_act: relu in tangent space at origin
  z[0] = 0.f;
  #pragma unroll
  for (int k = 1; k < HD; ++k) z[k] = fmaxf(z[k], 0.f);
  float o[HD];
  expmap0_proj(z, o);

  float* orow = out + (size_t)i * HD;
  #pragma unroll
  for (int k = 0; k < HD; ++k) orow[k] = o[k];
}

// ---------------------------------------------------------------------------
extern "C" void kernel_launch(void* const* d_in, const int* in_sizes, int n_in,
                              void* d_out, int out_size, void* d_ws, size_t ws_size,
                              hipStream_t stream) {
  (void)in_sizes; (void)n_in; (void)out_size; (void)ws_size;
  const float* x   = (const float*)d_in[0];
  const float* adj = (const float*)d_in[1];
  const float* W1  = (const float*)d_in[2];
  const float* b1  = (const float*)d_in[3];
  const float* W2  = (const float*)d_in[4];
  const float* b2  = (const float*)d_in[5];
  float* out = (float*)d_out;

  // ws layout: TbT (80 x 16384 bf16 = 2.62 MB) | partial (4 x 16384 x 80 f32)
  __bf16* TbT = (__bf16*)d_ws;
  float* partial = (float*)((char*)d_ws + (size_t)NTB * LDT * sizeof(__bf16));

  const int rows_blocks = HN / 256;                       // 64
  const int agg_blocks = (HN / 16) * SPLITK / 8;          // 4096 waves / 8 = 512

  // ---- layer 1 ----
  hgcn_pre_layer<<<rows_blocks, 256, 0, stream>>>(x, W1, b1, TbT, 1);
  hgcn_agg_wmma<<<agg_blocks, 256, 0, stream>>>(adj, TbT, partial);
  hgcn_post_agg<<<rows_blocks, 256, 0, stream>>>(partial, out);

  // ---- layer 2 (input = h1 just written to d_out) ----
  hgcn_pre_layer<<<rows_blocks, 256, 0, stream>>>(out, W2, b2, TbT, 0);
  hgcn_agg_wmma<<<agg_blocks, 256, 0, stream>>>(adj, TbT, partial);
  hgcn_post_agg<<<rows_blocks, 256, 0, stream>>>(partial, out + (size_t)HN * HD);
}
